// RecursiveReasoningTransformer_37864431681846
// MI455X (gfx1250) — matrix operand order repaired and checked by reading, not compiled, not run on hardware
//
#include <hip/hip_runtime.h>
#include <hip/hip_bf16.h>

// RecursiveReasoningTransformer forward for MI455X (gfx1250, wave32, WMMA).
// All GEMMs and both attention stages run through v_wmma_f32_16x16x32_f16
// (f16 inputs staged via LDS, f32 accumulation).  v2: batched global loads
// (single s_wait_loadcnt per tile) + double-buffered LDS tiles so tile t+1's
// HBM traffic overlaps tile t's WMMA work; flash inner loop reduced to one
// barrier per key tile.

typedef __attribute__((ext_vector_type(16))) _Float16 v16h;
typedef __attribute__((ext_vector_type(8)))  _Float16 v8h;
typedef __attribute__((ext_vector_type(8)))  float    v8f;

// ---------------------------------------------------------------------------
// Fragment loaders from LDS (f16), matching CDNA5 WMMA VGPR layouts.
//
// A operand (16xK rows=M): lane<16 -> row=lane, K {kb..kb+7} U {kb+16..kb+23}
// with kb=0; lanes 16..31 -> same rows, kb=8.
__device__ __forceinline__ v16h ld_fragA(const _Float16* base, int stride) {
  const int lane = threadIdx.x & 31;
  const _Float16* p = base + (lane & 15) * stride + ((lane >> 4) << 3);
  v8h lo = *(const v8h*)p;         // K kb..kb+7
  v8h hi = *(const v8h*)(p + 16);  // K kb+16..kb+23
  v16h f;
#pragma unroll
  for (int i = 0; i < 8; ++i) { f[i] = lo[i]; f[i + 8] = hi[i]; }
  return f;
}

// B operand (Kx16, cols=N), column-major per lane: lane<16 -> col=lane,
// K {kb..kb+7} U {kb+16..kb+23} with kb=0; lanes 16..31 -> same cols, kb=8.
__device__ __forceinline__ v16h ld_fragB(const _Float16* base, int stride) {
  const int lane = threadIdx.x & 31;
  const _Float16* p = base + (lane & 15) * stride + ((lane >> 4) << 3);
  v8h lo = *(const v8h*)p;         // K kb..kb+7
  v8h hi = *(const v8h*)(p + 16);  // K kb+16..kb+23
  v16h f;
#pragma unroll
  for (int i = 0; i < 8; ++i) { f[i] = lo[i]; f[i + 8] = hi[i]; }
  return f;
}

// ---------------------------------------------------------------------------
// GEMM + bias: C[M,N] = A[M,K] @ W[K,N] + b[N].  M%128==0, N%128==0, K%32==0.
// 256 threads = 8 waves; 128x128 tile; wave grid 4(M) x 2(N); each wave owns
// a 32x64 patch = 2x4 WMMA subtiles.  Double-buffered LDS K-steps.
__global__ __launch_bounds__(256)
void gemm_bias_kernel(const float* __restrict__ A, const float* __restrict__ W,
                      const float* __restrict__ bias, float* __restrict__ C,
                      int M, int K, int N) {
  __shared__ __align__(16) _Float16 lsA[2][128 * 40];  // [m][k], 32 K + 8 pad
  __shared__ __align__(16) _Float16 lsB[2][128 * 40];  // [n][k] (B column-major)

  const int tid  = threadIdx.x;
  const int lane = tid & 31, wid = tid >> 5;
  const int wm = wid & 3, wn = wid >> 2;
  const int bm = blockIdx.x * 128, bn = blockIdx.y * 128;
  const int r16 = lane & 15, hh = lane >> 4;

  float4 aR[4], bR[4];
  // Issue all 8 global_load_b128 for one K-step before any use (1 wait).
  auto loadTile = [&](int kt) {
#pragma unroll
    for (int it = 0; it < 4; ++it) {
      int f4 = tid + it * 256;
      int r  = f4 >> 3, c = (f4 & 7) << 2;
      aR[it] = *(const float4*)(A + (size_t)(bm + r) * K + kt * 32 + c);
      int kr = f4 >> 5, c2 = (f4 & 31) << 2;
      bR[it] = *(const float4*)(W + (size_t)(kt * 32 + kr) * N + bn + c2);
    }
  };
  auto storeTile = [&](int buf) {
#pragma unroll
    for (int it = 0; it < 4; ++it) {
      int f4 = tid + it * 256;
      int r  = f4 >> 3, c = (f4 & 7) << 2;
      _Float16* d = &lsA[buf][r * 40 + c];
      d[0] = (_Float16)aR[it].x; d[1] = (_Float16)aR[it].y;
      d[2] = (_Float16)aR[it].z; d[3] = (_Float16)aR[it].w;
      int kr = f4 >> 5, c2 = (f4 & 31) << 2;
      lsB[buf][(c2 + 0) * 40 + kr] = (_Float16)bR[it].x;
      lsB[buf][(c2 + 1) * 40 + kr] = (_Float16)bR[it].y;
      lsB[buf][(c2 + 2) * 40 + kr] = (_Float16)bR[it].z;
      lsB[buf][(c2 + 3) * 40 + kr] = (_Float16)bR[it].w;
    }
  };

  v8f vzero = {};
  v8f acc[2][4];
#pragma unroll
  for (int i = 0; i < 2; ++i)
#pragma unroll
    for (int j = 0; j < 4; ++j) acc[i][j] = vzero;

  const int nK = K / 32;
  loadTile(0);
  storeTile(0);
  __syncthreads();

  for (int kt = 0; kt < nK; ++kt) {
    const int cur = kt & 1;
    const bool more = (kt + 1) < nK;
    if (more) loadTile(kt + 1);  // loads in flight while WMMAs run below

    v16h af[2], bf[4];
#pragma unroll
    for (int i = 0; i < 2; ++i)
      af[i] = ld_fragA(&lsA[cur][(wm * 32 + i * 16) * 40], 40);
#pragma unroll
    for (int j = 0; j < 4; ++j)
      bf[j] = ld_fragB(&lsB[cur][(wn * 64 + j * 16) * 40], 40);
#pragma unroll
    for (int i = 0; i < 2; ++i)
#pragma unroll
      for (int j = 0; j < 4; ++j)
        acc[i][j] = __builtin_amdgcn_wmma_f32_16x16x32_f16(
            false, af[i], false, bf[j], (short)0, acc[i][j], false, false);

    if (more) storeTile(cur ^ 1);  // write alternate buffer (read last at kt-1)
    __syncthreads();
  }

  // C layout: lane<16 -> N=lane, M=vgpr; lanes 16..31 -> N=lane-16, M=vgpr+8.
#pragma unroll
  for (int i = 0; i < 2; ++i)
#pragma unroll
    for (int j = 0; j < 4; ++j) {
      int col  = bn + wn * 64 + j * 16 + r16;
      float bc = bias[col];
#pragma unroll
      for (int r = 0; r < 8; ++r) {
        int row = bm + wm * 32 + i * 16 + r + hh * 8;
        C[(size_t)row * N + col] = acc[i][j][r] + bc;
      }
    }
}

// ---------------------------------------------------------------------------
// Flash attention, one head per blockIdx.y, 64 query rows per blockIdx.x.
// 128 threads = 4 waves, each wave owns 16 query rows (online softmax).
// Double-buffered K/V tiles; one barrier per key tile.
__global__ __launch_bounds__(128)
void flash_attn_kernel(const float* __restrict__ Q, const float* __restrict__ Kb,
                       const float* __restrict__ Vb, float* __restrict__ O,
                       int S, int Hdim, float scale) {
  __shared__ __align__(16) _Float16 lsQ[64 * 72];         // [qrow][d]
  __shared__ __align__(16) _Float16 lsK[2][32 * 72];      // [key][d]
  __shared__ __align__(16) _Float16 lsV[2][64 * 40];      // [d][key]
  __shared__ __align__(16) _Float16 lsP[4 * 16 * 40];     // per-wave exp(P)

  const int tid  = threadIdx.x;
  const int lane = tid & 31, wid = tid >> 5;
  const int r16 = lane & 15, hh = lane >> 4;
  const int qbase = blockIdx.x * 64;
  const int hcol  = blockIdx.y * 64;

  float4 kR[4], vR[4];
  auto loadKV = [&](int t) {
#pragma unroll
    for (int it = 0; it < 4; ++it) {
      int f4 = tid + it * 128;
      int r  = f4 >> 4, c = (f4 & 15) << 2;
      kR[it] = *(const float4*)(Kb + (size_t)(t * 32 + r) * Hdim + hcol + c);
      vR[it] = *(const float4*)(Vb + (size_t)(t * 32 + r) * Hdim + hcol + c);
    }
  };
  auto storeKV = [&](int buf) {
#pragma unroll
    for (int it = 0; it < 4; ++it) {
      int f4 = tid + it * 128;
      int r  = f4 >> 4, c = (f4 & 15) << 2;
      _Float16* d = &lsK[buf][r * 72 + c];
      d[0] = (_Float16)kR[it].x; d[1] = (_Float16)kR[it].y;
      d[2] = (_Float16)kR[it].z; d[3] = (_Float16)kR[it].w;
      lsV[buf][(c + 0) * 40 + r] = (_Float16)vR[it].x;
      lsV[buf][(c + 1) * 40 + r] = (_Float16)vR[it].y;
      lsV[buf][(c + 2) * 40 + r] = (_Float16)vR[it].z;
      lsV[buf][(c + 3) * 40 + r] = (_Float16)vR[it].w;
    }
  };

  // Stage Q (64x64 fp32 -> f16) and first K/V tile.
#pragma unroll
  for (int it = 0; it < 8; ++it) {
    int f4 = tid + it * 128;
    int r  = f4 >> 4, c = (f4 & 15) << 2;
    const float4 qv = *(const float4*)(Q + (size_t)(qbase + r) * Hdim + hcol + c);
    _Float16* d = lsQ + r * 72 + c;
    d[0] = (_Float16)qv.x; d[1] = (_Float16)qv.y;
    d[2] = (_Float16)qv.z; d[3] = (_Float16)qv.w;
  }
  loadKV(0);
  storeKV(0);
  __syncthreads();

  v16h aq[2];
#pragma unroll
  for (int ks = 0; ks < 2; ++ks)
    aq[ks] = ld_fragA(lsQ + (wid * 16) * 72 + ks * 32, 72);

  float m[8], l[8];
  v8f vzero = {};
  v8f o[4];
#pragma unroll
  for (int r = 0; r < 8; ++r) { m[r] = -1e30f; l[r] = 0.0f; }
#pragma unroll
  for (int f = 0; f < 4; ++f) o[f] = vzero;

  _Float16* myP = lsP + wid * 16 * 40;
  const int nT = S / 32;
  for (int t = 0; t < nT; ++t) {
    const int cur = t & 1;
    const bool more = (t + 1) < nT;
    if (more) loadKV(t + 1);  // overlap next tile's HBM traffic with compute

    // S = Q @ K^T for this wave's 16 rows x 32 keys (two 16x16 C frags).
    v8f s0 = vzero, s1 = vzero;
#pragma unroll
    for (int ks = 0; ks < 2; ++ks) {
      v16h bk0 = ld_fragB(&lsK[cur][0  * 72 + ks * 32], 72);
      v16h bk1 = ld_fragB(&lsK[cur][16 * 72 + ks * 32], 72);
      s0 = __builtin_amdgcn_wmma_f32_16x16x32_f16(false, aq[ks], false, bk0,
                                                  (short)0, s0, false, false);
      s1 = __builtin_amdgcn_wmma_f32_16x16x32_f16(false, aq[ks], false, bk1,
                                                  (short)0, s1, false, false);
    }

    // Online softmax: row stats live across the 16-lane half of the C layout.
    float corr[8];
#pragma unroll
    for (int r = 0; r < 8; ++r) {
      float x = fmaxf(s0[r], s1[r]) * scale;
      x = fmaxf(x, __shfl_xor(x, 1));
      x = fmaxf(x, __shfl_xor(x, 2));
      x = fmaxf(x, __shfl_xor(x, 4));
      x = fmaxf(x, __shfl_xor(x, 8));
      float mn = fmaxf(m[r], x);
      float cr = __expf(m[r] - mn);
      float p0 = __expf(s0[r] * scale - mn);
      float p1 = __expf(s1[r] * scale - mn);
      float rs = p0 + p1;
      rs += __shfl_xor(rs, 1);
      rs += __shfl_xor(rs, 2);
      rs += __shfl_xor(rs, 4);
      rs += __shfl_xor(rs, 8);
      m[r] = mn;
      l[r] = l[r] * cr + rs;
      corr[r] = cr;
      _Float16* pp = myP + (r + hh * 8) * 40 + r16;
      pp[0]  = (_Float16)p0;   // keys 0..15 subtile
      pp[16] = (_Float16)p1;   // keys 16..31 subtile
    }
#pragma unroll
    for (int f = 0; f < 4; ++f)
#pragma unroll
      for (int r = 0; r < 8; ++r) o[f][r] *= corr[r];

    // exp(P) round-trip is per-wave: DS ops from one wave stay in order, so a
    // compiler fence (no s_barrier) is enough to order store->load on lsP.
    asm volatile("" ::: "memory");

    // O += P @ V  (A = exp(P) 16x32, B = V 32x16 per d-subtile).
    v16h ap = ld_fragA(myP, 40);
#pragma unroll
    for (int f = 0; f < 4; ++f) {
      v16h bv = ld_fragB(&lsV[cur][(f * 16) * 40], 40);
      o[f] = __builtin_amdgcn_wmma_f32_16x16x32_f16(false, ap, false, bv,
                                                    (short)0, o[f], false, false);
    }

    if (more) storeKV(cur ^ 1);  // alternate buffer, last read at t-1
    __syncthreads();
  }

  float inv[8];
#pragma unroll
  for (int r = 0; r < 8; ++r) inv[r] = 1.0f / l[r];
#pragma unroll
  for (int f = 0; f < 4; ++f)
#pragma unroll
    for (int r = 0; r < 8; ++r) {
      int row = qbase + wid * 16 + r + hh * 8;
      O[(size_t)row * Hdim + hcol + f * 16 + r16] = o[f][r] * inv[r];
    }
}

// ---------------------------------------------------------------------------
// cat[:, :H] = output ; cat[:, H:] = 0.5*(rc0 + rc1)   (rmean lives in cat)
__global__ void concat_mean_kernel(const float* __restrict__ outp,
                                   const float* __restrict__ rc0,
                                   const float* __restrict__ rc1,
                                   float* __restrict__ cat, int SH, int Hdim) {
  int i = blockIdx.x * blockDim.x + threadIdx.x;
  if (i >= SH) return;
  int row = i / Hdim, col = i - row * Hdim;
  float rm = 0.5f * (rc0[i] + rc1[i]);
  cat[(size_t)row * (2 * Hdim) + col] = outp[i];
  cat[(size_t)row * (2 * Hdim) + Hdim + col] = rm;
}

// out = output + sigmoid(g) * (rmean - output)
__global__ void gate_out_kernel(const float* __restrict__ outp,
                                const float* __restrict__ cat,
                                const float* __restrict__ g,
                                float* __restrict__ dst, int SH, int Hdim) {
  int i = blockIdx.x * blockDim.x + threadIdx.x;
  if (i >= SH) return;
  int row = i / Hdim, col = i - row * Hdim;
  float rm = cat[(size_t)row * (2 * Hdim) + Hdim + col];
  float s  = 1.0f / (1.0f + __expf(-g[i]));
  dst[i] = outp[i] + s * (rm - outp[i]);
}

// ---------------------------------------------------------------------------
extern "C" void kernel_launch(void* const* d_in, const int* in_sizes, int n_in,
                              void* d_out, int out_size, void* d_ws, size_t ws_size,
                              hipStream_t stream) {
  (void)in_sizes; (void)n_in; (void)out_size; (void)ws_size;
  const int S = 2048, H = 1024, NH = 16;
  const float SCALE = 0.125f;  // 1/sqrt(64)
  const size_t SH = (size_t)S * H;

  const float* hs   = (const float*)d_in[0];
  const float* past = (const float*)d_in[1];   // [2, S, H] flat
  const float* Wq  = (const float*)d_in[2];  const float* bq  = (const float*)d_in[3];
  const float* Wk  = (const float*)d_in[4];  const float* bk  = (const float*)d_in[5];
  const float* Wv  = (const float*)d_in[6];  const float* bv  = (const float*)d_in[7];
  const float* Wo  = (const float*)d_in[8];  const float* bo  = (const float*)d_in[9];
  const float* Wrq = (const float*)d_in[10]; const float* brq = (const float*)d_in[11];
  const float* Wrk = (const float*)d_in[12]; const float* brk = (const float*)d_in[13];
  const float* Wrv = (const float*)d_in[14]; const float* brv = (const float*)d_in[15];
  const float* Wg  = (const float*)d_in[16]; const float* bg  = (const float*)d_in[17];

  // Workspace layout (floats), 10*SH total = 80 MB, with buffer reuse:
  float* ws   = (float*)d_ws;
  float* q    = ws + 0 * SH;           // q, later rq
  float* kbuf = ws + 1 * SH;           // k, later rk[2S,H] (spans 1SH..3SH)
  float* vbuf = ws + 2 * SH;           // v
  float* ctx  = ws + 3 * SH;           // ctx, later rv[2S,H] (spans 3SH..5SH)
  float* outp = ws + 5 * SH;           // output (live to the end)
  float* rc0  = ws + 6 * SH;           // rctx p=0, later gate logits
  float* rc1  = ws + 7 * SH;           // rctx p=1
  float* cat  = ws + 8 * SH;           // [S, 2H]
  float* rq = q;
  float* rk = kbuf;
  float* rv = ctx;
  float* g  = rc0;

  dim3 gemmGrid(S / 128, H / 128);
  dim3 gemmGrid2(2 * S / 128, H / 128);
  dim3 attnGrid(S / 64, NH);

  // --- self attention ---
  gemm_bias_kernel<<<gemmGrid, 256, 0, stream>>>(hs, Wq, bq, q,    S, H, H);
  gemm_bias_kernel<<<gemmGrid, 256, 0, stream>>>(hs, Wk, bk, kbuf, S, H, H);
  gemm_bias_kernel<<<gemmGrid, 256, 0, stream>>>(hs, Wv, bv, vbuf, S, H, H);
  flash_attn_kernel<<<attnGrid, 128, 0, stream>>>(q, kbuf, vbuf, ctx, S, H, SCALE);
  gemm_bias_kernel<<<gemmGrid, 256, 0, stream>>>(ctx, Wo, bo, outp, S, H, H);

  // --- recursive attention over past states (P=2) ---
  gemm_bias_kernel<<<gemmGrid,  256, 0, stream>>>(outp, Wrq, brq, rq, S, H, H);
  gemm_bias_kernel<<<gemmGrid2, 256, 0, stream>>>(past, Wrk, brk, rk, 2 * S, H, H);
  gemm_bias_kernel<<<gemmGrid2, 256, 0, stream>>>(past, Wrv, brv, rv, 2 * S, H, H);
  flash_attn_kernel<<<attnGrid, 128, 0, stream>>>(rq, rk,      rv,      rc0, S, H, SCALE);
  flash_attn_kernel<<<attnGrid, 128, 0, stream>>>(rq, rk + SH, rv + SH, rc1, S, H, SCALE);

  // --- gate ---
  int ewBlocks = (int)((SH + 255) / 256);
  concat_mean_kernel<<<ewBlocks, 256, 0, stream>>>(outp, rc0, rc1, cat, (int)SH, H);
  gemm_bias_kernel<<<gemmGrid, 256, 0, stream>>>(cat, Wg, bg, g, S, 2 * H, H);
  gate_out_kernel<<<ewBlocks, 256, 0, stream>>>(outp, cat, g, (float*)d_out, (int)SH, H);
}
